// RiskPredictionGNN_1700807050070
// MI455X (gfx1250) — matrix operand order repaired and checked by reading, not compiled, not run on hardware
//
#include <hip/hip_runtime.h>
#include <cstdint>

#define HID 256

typedef __attribute__((ext_vector_type(16))) __bf16 v16bf;
typedef __attribute__((ext_vector_type(8)))  float  v8f;

static __device__ __forceinline__ unsigned short f2bf(float f) {
  unsigned int u = __float_as_uint(f);
  unsigned int r = (u + 0x7FFFu + ((u >> 16) & 1u)) >> 16;  // RNE
  return (unsigned short)r;
}
static __device__ __forceinline__ float bf2f(unsigned short s) {
  return __uint_as_float(((unsigned int)s) << 16);
}

// Packed WMMA-fragment index for an M x 256 bf16 matrix:
//   frag element e of lane L of k-chunk kc8 of row-tile mtile sits at
//   ((mtile*8 + kc8)*32 + L)*16 + e
// Forward mapping (ISA 16-bit A layout): lane L = halfsel*16 + (m%16),
//   k = (e>>3)*16 + halfsel*8 + ((e>>1)&3)*2 + (e&1)
static __device__ __forceinline__ size_t pack_idx(int node, int f) {
  int mtile = node >> 4, laneM = node & 15;
  int kc8 = f >> 5, kin = f & 31;
  int g = kin >> 4, rem = kin & 15;
  int halfsel = rem >> 3, w8 = rem & 7;
  int j = w8 >> 1, par = w8 & 1;
  int e = ((g << 2) + j) * 2 + par;
  int lane = (halfsel << 4) + laneM;
  return (((size_t)(mtile * 8 + kc8) * 32 + lane) << 4) + e;
}

// ---------------- elementwise helpers ----------------

__global__ void k_fill_f32(float* p, float v, int n) {
  int i = blockIdx.x * blockDim.x + threadIdx.x;
  if (i < n) p[i] = v;
}

__global__ void k_deg_edges(const long long* dst, float* deg, int E) {
  int i = blockIdx.x * blockDim.x + threadIdx.x;
  if (i < E) atomicAdd(&deg[(int)dst[i]], 1.0f);
}

__global__ void k_rsqrt_inplace(float* p, int n) {
  int i = blockIdx.x * blockDim.x + threadIdx.x;
  if (i < n) { float d = p[i]; p[i] = (d > 0.0f) ? rsqrtf(d) : 0.0f; }
}

// cast f32 row-major -> packed bf16 fragments
__global__ void k_cast_pack(const float* src, unsigned short* Apack, int total) {
  int i = blockIdx.x * blockDim.x + threadIdx.x;
  if (i >= total) return;
  Apack[pack_idx(i >> 8, i & 255)] = f2bf(src[i]);
}

// pack weight matrix (256x256 f32, row-major K x N) into B fragments:
// [ntile][kc8][lane][e], n = ntile*16 + lane%16, k = kc8*32 + (e/8)*16 + (lane/16)*8 + e%8
__global__ void k_pack_w(const float* W, unsigned short* Bpack) {
  int i = blockIdx.x * blockDim.x + threadIdx.x;   // 65536
  if (i >= HID * HID) return;
  int e = i & 15, lane = (i >> 4) & 31, kc8 = (i >> 9) & 7, ntile = i >> 12;
  int n = ntile * 16 + (lane & 15);
  int k = kc8 * 32 + ((e >> 3) << 4) + ((lane >> 4) << 3) + (e & 7);
  Bpack[i] = f2bf(W[(size_t)k * HID + n]);
}

// hi_in[i][f] = relu(x[i][255] * Wm1[f] + bm1[f]) -> packed bf16
__global__ void k_impact_in_pack(const float* x, const float* Wm1, const float* bm1,
                                 unsigned short* Apack, int total) {
  int i = blockIdx.x * blockDim.x + threadIdx.x;
  if (i >= total) return;
  int node = i >> 8, f = i & 255;
  float v = fmaxf(fmaf(x[node * HID + 255], Wm1[f], bm1[f]), 0.0f);
  Apack[pack_idx(node, f)] = f2bf(v);
}

// HI[i][f] = bf16(ACC[i][f] + bm2[f])   (row-major; consumed by classifier)
__global__ void k_hi_final(const float* ACC, const float* bm2, unsigned short* HI, int total) {
  int i = blockIdx.x * blockDim.x + threadIdx.x;
  if (i < total) HI[i] = f2bf(ACC[i] + bm2[i & 255]);
}

// y = xw * dinv[src]; AGG initialized with self-loop y (one fused pass)
__global__ void k_scale_and_seed(float* ACC, float* AGG, const float* dinv, int total) {
  int i = blockIdx.x * blockDim.x + threadIdx.x;
  if (i >= total) return;
  float y = ACC[i] * dinv[i >> 8];
  ACC[i] = y;
  AGG[i] = y;
}

// one wave per edge: AGG[dst] += y[src]  (256 floats, 8 per lane, fp32 atomics; L2-resident)
__global__ void k_agg_edges(const long long* src, const long long* dst,
                            const float* Y, float* AGG, int E) {
  int gid = blockIdx.x * blockDim.x + threadIdx.x;
  int e = gid >> 5;
  int lane = threadIdx.x & 31;
  if (e >= E) return;
  int s = (int)src[e], d = (int)dst[e];
  const float* y = Y + (size_t)s * HID;
  float* o = AGG + (size_t)d * HID;
#pragma unroll
  for (int j = 0; j < 8; j++) {
    int f = lane + 32 * j;
    atomicAdd(o + f, y[f]);
  }
}

// H[i][f] = bf16(relu(dinv[i]*AGG[i][f] + b[f])) -> packed bf16 fragments
__global__ void k_layer_final_pack(const float* AGG, const float* dinv, const float* bias,
                                   unsigned short* Apack, int total) {
  int i = blockIdx.x * blockDim.x + threadIdx.x;
  if (i >= total) return;
  int node = i >> 8, f = i & 255;
  Apack[pack_idx(node, f)] = f2bf(fmaxf(fmaf(dinv[node], AGG[i], bias[f]), 0.0f));
}

// ---------------- WMMA GEMM on packed fragments, double-buffered ----------------
// C[M x 256] = A[M x 256](bf16 packed) @ B[256 x 256](bf16 packed)
// One wave computes a 16x64 strip. K chunk k+1's fragments (2 A + 8 B b128 loads)
// are issued BEFORE chunk k's 4 WMMAs, so each WMMA waits only on loads issued
// a full chunk earlier (in-wave latency hiding).
__global__ __launch_bounds__(32) void k_gemm_bf16_wmma(const uint4* __restrict__ Apack,
                                                       const uint4* __restrict__ Bpack,
                                                       float* __restrict__ C, int M) {
  const int lane = threadIdx.x;                       // 0..31
  const int mtile = blockIdx.x;
  const int nt4 = blockIdx.y;                         // 4 n-tiles of 16
  const int m0 = mtile * 16;
  const int halfsel = (lane >> 4) & 1;

  v8f acc[4] = {};

  // per-lane fragment bases: 2 uint4 (32 B) per lane per chunk; chunk stride = 64 uint4
  const uint4* aptr0 = Apack + ((size_t)(mtile * 8) * 32 + lane) * 2;
  const uint4* bbase = Bpack + ((size_t)(nt4 * 4) * 8 * 32 + lane) * 2;

  union Frag { v16bf v; uint4 q[2]; };
  Frag a[2];
  Frag b[2][4];

  // prologue: chunk 0
  a[0].q[0] = aptr0[0];
  a[0].q[1] = aptr0[1];
#pragma unroll
  for (int t = 0; t < 4; t++) {
    const uint4* bp = bbase + ((size_t)t * 8) * 64;
    b[0][t].q[0] = bp[0];
    b[0][t].q[1] = bp[1];
  }

#pragma unroll
  for (int kc8 = 0; kc8 < 8; kc8++) {
    const int cur = kc8 & 1, nxt = cur ^ 1;
    if (kc8 < 7) {
      // issue next chunk's loads before this chunk's WMMAs
      const uint4* ap = aptr0 + (size_t)(kc8 + 1) * 64;
      a[nxt].q[0] = ap[0];
      a[nxt].q[1] = ap[1];
#pragma unroll
      for (int t = 0; t < 4; t++) {
        const uint4* bp = bbase + ((size_t)t * 8 + (kc8 + 1)) * 64;
        b[nxt][t].q[0] = bp[0];
        b[nxt][t].q[1] = bp[1];
      }
      if (kc8 < 6) __builtin_prefetch(aptr0 + (size_t)(kc8 + 2) * 64, 0, 1);  // global_prefetch_b8
    }
#pragma unroll
    for (int t = 0; t < 4; t++) {
      acc[t] = __builtin_amdgcn_wmma_f32_16x16x32_bf16(
          /*neg_a=*/false, a[cur].v, /*neg_b=*/false, b[cur][t].v,
          /*c_mod=*/(short)0, acc[t], /*reuse_a=*/false, /*reuse_b=*/false);
    }
  }

  // C/D layout: VGPR i -> row m0 + i + halfsel*8, column = lane%16
  if (m0 + 15 < M) {  // fast path (always taken for N % 16 == 0)
#pragma unroll
    for (int t = 0; t < 4; t++) {
      const int ncol = (nt4 * 4 + t) * 16 + (lane & 15);
#pragma unroll
      for (int i = 0; i < 8; i++) {
        int m = m0 + i + (halfsel << 3);
        C[(size_t)m * HID + ncol] = acc[t][i];
      }
    }
  } else {
#pragma unroll
    for (int t = 0; t < 4; t++) {
      const int ncol = (nt4 * 4 + t) * 16 + (lane & 15);
#pragma unroll
      for (int i = 0; i < 8; i++) {
        int m = m0 + i + (halfsel << 3);
        if (m < M) C[(size_t)m * HID + ncol] = acc[t][i];
      }
    }
  }
}

// ---------------- classifier: wave per node, logits = [h2, hi] @ Wc + bc, log_softmax ----
// H2 is in packed-fragment layout; with f = lane + 32*j the pack coords are per-lane constants.
__global__ void k_classifier(const unsigned short* H2pack, const unsigned short* HI,
                             const float* Wc, const float* bc, float* out, int N) {
  int gid = blockIdx.x * blockDim.x + threadIdx.x;
  int node = gid >> 5;
  int lane = threadIdx.x & 31;
  if (node >= N) return;

  // decompose kin = lane once
  int g = lane >> 4, rem = lane & 15;
  int halfsel = rem >> 3, w8 = rem & 7;
  int e = ((g << 2) + (w8 >> 1)) * 2 + (w8 & 1);
  int laneIdx = (halfsel << 4) + (node & 15);
  size_t mbase = (size_t)(node >> 4) * 8;

  float lg0 = 0.f, lg1 = 0.f, lg2 = 0.f, lg3 = 0.f;
#pragma unroll
  for (int j = 0; j < 8; j++) {
    int f = lane + 32 * j;                       // kc8 == j
    float h  = bf2f(H2pack[(((mbase + j) * 32 + laneIdx) << 4) + e]);
    float hi = bf2f(HI[(size_t)node * HID + f]);
    const float* w0 = Wc + (size_t)f * 4;
    const float* w1 = Wc + (size_t)(HID + f) * 4;
    lg0 += h * w0[0] + hi * w1[0];
    lg1 += h * w0[1] + hi * w1[1];
    lg2 += h * w0[2] + hi * w1[2];
    lg3 += h * w0[3] + hi * w1[3];
  }
#pragma unroll
  for (int off = 16; off > 0; off >>= 1) {
    lg0 += __shfl_down(lg0, off);
    lg1 += __shfl_down(lg1, off);
    lg2 += __shfl_down(lg2, off);
    lg3 += __shfl_down(lg3, off);
  }
  if (lane == 0) {
    float l0 = lg0 + bc[0], l1 = lg1 + bc[1], l2 = lg2 + bc[2], l3 = lg3 + bc[3];
    float mx = fmaxf(fmaxf(l0, l1), fmaxf(l2, l3));
    float se = expf(l0 - mx) + expf(l1 - mx) + expf(l2 - mx) + expf(l3 - mx);
    float ls = mx + logf(se);
    float* o = out + (size_t)node * 4;
    o[0] = l0 - ls; o[1] = l1 - ls; o[2] = l2 - ls; o[3] = l3 - ls;
  }
}

// ---------------- host-side orchestration ----------------

extern "C" void kernel_launch(void* const* d_in, const int* in_sizes, int n_in,
                              void* d_out, int out_size, void* d_ws, size_t ws_size,
                              hipStream_t stream) {
  const float*     x   = (const float*)d_in[0];
  const long long* ei  = (const long long*)d_in[1];   // int64 (2, E)
  const float*     W1  = (const float*)d_in[2];
  const float*     b1  = (const float*)d_in[3];
  const float*     W2  = (const float*)d_in[4];
  const float*     b2  = (const float*)d_in[5];
  const float*     Wm1 = (const float*)d_in[6];
  const float*     bm1 = (const float*)d_in[7];
  const float*     Wm2 = (const float*)d_in[8];
  const float*     bm2 = (const float*)d_in[9];
  const float*     Wc  = (const float*)d_in[10];
  const float*     bc  = (const float*)d_in[11];
  float* out = (float*)d_out;

  const int Nn = in_sizes[0] / HID;             // 100000 nodes
  const int E  = in_sizes[1] / 2;               // 1.6M edges
  const long long* esrc = ei;
  const long long* edst = ei + E;
  const int Mtiles = (Nn + 15) / 16;

  // workspace layout
  char* ws = (char*)d_ws;
  auto al = [](size_t v) { return (v + 255) & ~(size_t)255; };
  const size_t total = (size_t)Nn * HID;
  const size_t packElems = (size_t)Mtiles * 8 * 32 * 16;   // padded packed A
  size_t off = 0;
  float*          dinv  = (float*)(ws + off);          off = al(off + (size_t)Nn * 4);
  unsigned short* Apack = (unsigned short*)(ws + off); off = al(off + packElems * 2);
  float*          ACC   = (float*)(ws + off);          off = al(off + total * 4);
  float*          AGG   = (float*)(ws + off);          off = al(off + total * 4);
  unsigned short* HIb   = (unsigned short*)(ws + off); off = al(off + total * 2);
  unsigned short* Bpack = (unsigned short*)(ws + off); off = al(off + (size_t)HID * HID * 2);
  (void)ws_size; (void)n_in; (void)out_size;

  const dim3 b256(256);
  const int gElem = (int)((total + 255) / 256);
  const int gN    = (Nn + 255) / 256;
  const int gE    = (E + 255) / 256;
  const int gEw   = (int)(((size_t)E * 32 + 255) / 256);   // wave-per-edge
  const int gW    = (HID * HID + 255) / 256;
  const int gCls  = (int)(((size_t)Nn * 32 + 255) / 256);  // wave-per-node
  const dim3 gGemm(Mtiles, HID / 64);
  const dim3 b32(32);

  // ---- symmetric normalization: dinv = rsqrt(1 + indegree) ----
  k_fill_f32<<<gN, b256, 0, stream>>>(dinv, 1.0f, Nn);
  k_deg_edges<<<gE, b256, 0, stream>>>(edst, dinv, E);
  k_rsqrt_inplace<<<gN, b256, 0, stream>>>(dinv, Nn);

  // ---- impact branch: HI = bf16(relu(x[:,255]*Wm1 + bm1) @ Wm2 + bm2) ----
  k_impact_in_pack<<<gElem, b256, 0, stream>>>(x, Wm1, bm1, Apack, (int)total);
  k_pack_w<<<gW, b256, 0, stream>>>(Wm2, Bpack);
  k_gemm_bf16_wmma<<<gGemm, b32, 0, stream>>>((const uint4*)Apack, (const uint4*)Bpack, ACC, Nn);
  k_hi_final<<<gElem, b256, 0, stream>>>(ACC, bm2, HIb, (int)total);

  // ---- GCN layer 1 ----
  k_cast_pack<<<gElem, b256, 0, stream>>>(x, Apack, (int)total);
  k_pack_w<<<gW, b256, 0, stream>>>(W1, Bpack);
  k_gemm_bf16_wmma<<<gGemm, b32, 0, stream>>>((const uint4*)Apack, (const uint4*)Bpack, ACC, Nn);
  k_scale_and_seed<<<gElem, b256, 0, stream>>>(ACC, AGG, dinv, (int)total);
  k_agg_edges<<<gEw, b256, 0, stream>>>(esrc, edst, ACC, AGG, E);
  k_layer_final_pack<<<gElem, b256, 0, stream>>>(AGG, dinv, b1, Apack, (int)total);

  // ---- GCN layer 2 ----
  k_pack_w<<<gW, b256, 0, stream>>>(W2, Bpack);
  k_gemm_bf16_wmma<<<gGemm, b32, 0, stream>>>((const uint4*)Apack, (const uint4*)Bpack, ACC, Nn);
  k_scale_and_seed<<<gElem, b256, 0, stream>>>(ACC, AGG, dinv, (int)total);
  k_agg_edges<<<gEw, b256, 0, stream>>>(esrc, edst, ACC, AGG, E);
  k_layer_final_pack<<<gElem, b256, 0, stream>>>(AGG, dinv, b2, Apack, (int)total);

  // ---- classifier + log_softmax ----
  k_classifier<<<gCls, b256, 0, stream>>>(Apack, HIb, Wc, bc, out, Nn);
}
